// GPTNeoSelfAttention_51324859187293
// MI455X (gfx1250) — matrix operand order repaired and checked by reading, not compiled, not run on hardware
//
#include <hip/hip_runtime.h>
#include <hip/hip_bf16.h>

// Problem constants (GPT-Neo self-attention reference)
#define BB 4
#define HH 16
#define SS 2048
#define DD 2048
#define DH 128
#define MM (BB * SS)  // 8192 rows
#define BHN (BB * HH) // 64 (batch*heads)

typedef __bf16 bf16_t;
typedef bf16_t v16bf __attribute__((ext_vector_type(16)));
typedef float v8f __attribute__((ext_vector_type(8)));
typedef unsigned u32;
typedef u32 v8u __attribute__((ext_vector_type(8)));

#define MINIT (-1.0e30f) // finite sentinel (avoids inf-inf NaN in merges)

// Round-to-nearest-even f32->bf16, two at a time, packed with one v_perm_b32.
__device__ __forceinline__ u32 pack2_bf16(float f0, float f1) {
  u32 u0 = __builtin_bit_cast(u32, f0);
  u32 u1 = __builtin_bit_cast(u32, f1);
  u0 += 0x7FFFu + ((u0 >> 16) & 1u);
  u1 += 0x7FFFu + ((u1 >> 16) & 1u);
  // result bytes {u1.b3,u1.b2,u0.b3,u0.b2} = {bf(f1), bf(f0)}
  return __builtin_amdgcn_perm(u1, u0, 0x07060302u);
}

__device__ __forceinline__ bf16_t f2bf(float f) {
  u32 u = __builtin_bit_cast(u32, f);
  u32 r = u + 0x7FFFu + ((u >> 16) & 1u);
  unsigned short h = (unsigned short)(r >> 16);
  return __builtin_bit_cast(bf16_t, h);
}

// ============================================================
// Kernel 0: bulk fp32 -> packed bf16 conversion (b128 in / b128 out).
// One thread converts 8 floats -> 4 packed u32.
// ============================================================
__global__ __launch_bounds__(256) void cvt_bf16_kernel(
    const float4* __restrict__ src, uint4* __restrict__ dst, int ngroups) {
  int i = blockIdx.x * 256 + threadIdx.x;
  if (i < ngroups) {
    float4 f0 = src[2 * i];
    float4 f1 = src[2 * i + 1];
    uint4 o;
    o.x = pack2_bf16(f0.x, f0.y);
    o.y = pack2_bf16(f0.z, f0.w);
    o.z = pack2_bf16(f1.x, f1.y);
    o.w = pack2_bf16(f1.z, f1.w);
    dst[i] = o;
  }
}

// ---- WMMA fragment loaders (layouts per CDNA5 ISA 7.12.2, wave32) ----
// A (16x32 bf16, MxK): lane l holds row M=l&15; half grp=l>>4.
//   VGPR v holds the K pair {base,base+1}, base = (v<4 ? 2v : 2v+8) + 8*grp.
//   Pairs are contiguous & 4B aligned -> one b32 load per VGPR (no repack).
__device__ __forceinline__ v16bf load_a_bf16(const bf16_t* p, int ld, int m0,
                                             int k0, int lane) {
  int grp = (lane >> 4) & 1;
  const u32* base =
      (const u32*)(p + (size_t)(m0 + (lane & 15)) * ld + k0 + grp * 8);
  v8u a;
#pragma unroll
  for (int v = 0; v < 8; ++v) a[v] = base[(v < 4) ? v : (v + 4)];
  return __builtin_bit_cast(v16bf, a);
}

// B (32x16 bf16, KxN) addressed through a row-major matrix W with
// Bop[k][n] = W[n0+n][k0+k]  (exactly nn.Linear's x @ W^T operand).
// Lane l: N = l&15 ; VGPR v holds K pair {base,base+1}, base = (v<4 ? 2v : 2v+8) + 8*grp -> b32.
__device__ __forceinline__ v16bf load_bT_bf16(const bf16_t* w, int ld, int n0,
                                              int k0, int lane) {
  int grp = (lane >> 4) & 1;
  const u32* base =
      (const u32*)(w + (size_t)(n0 + (lane & 15)) * ld + k0 + grp * 8);
  v8u b;
#pragma unroll
  for (int v = 0; v < 8; ++v) b[v] = base[(v < 4) ? v : (v + 4)];
  return __builtin_bit_cast(v16bf, b);
}

__device__ __forceinline__ v8f wmma_bf(v16bf a, v16bf b, v8f c) {
  // (neg_a, A, neg_b, B, c_mod, C, reuse_a, reuse_b)
  return __builtin_amdgcn_wmma_f32_16x16x32_bf16(false, a, false, b, (short)0,
                                                 c, false, false);
}

// ============================================================
// Kernel 1: projection GEMM  C[m,n] = sum_k X[m,k]*W[n,k]  (all bf16 in)
// vt_mode=1 stores V transposed per head: Vt[(b*H+h)*DH+dh][s].
// Block = 256 thr (8 waves as 2Mx4N); wave tile = 32x64 -> WG tile 64x256.
// ============================================================
__global__ __launch_bounds__(256) void gemm_qkv_kernel(
    const bf16_t* __restrict__ X, const bf16_t* __restrict__ W,
    bf16_t* __restrict__ out, int vt_mode) {
  int lane = threadIdx.x & 31;
  int wave = threadIdx.x >> 5;
  int wm = wave >> 2, wn = wave & 3;
  int m0 = blockIdx.y * 64 + wm * 32;
  int n0 = blockIdx.x * 256 + wn * 64;

  v8f acc[2][4];
#pragma unroll
  for (int i = 0; i < 2; ++i)
#pragma unroll
    for (int j = 0; j < 4; ++j) acc[i][j] = (v8f){};

  const bf16_t* aRow0 = X + (size_t)(m0 + (lane & 15)) * DD;
  const bf16_t* bRow0 = W + (size_t)(n0 + (lane & 15)) * DD;

  for (int k0 = 0; k0 < DD; k0 += 32) {
    if (k0 + 32 < DD) { // prefetch next k-slab (global_prefetch_b8)
      __builtin_prefetch(aRow0 + k0 + 32, 0, 3);
      __builtin_prefetch(bRow0 + k0 + 32, 0, 3);
    }
    v16bf a0 = load_a_bf16(X, DD, m0, k0, lane);
    v16bf a1 = load_a_bf16(X, DD, m0 + 16, k0, lane);
#pragma unroll
    for (int j = 0; j < 4; ++j) {
      v16bf bj = load_bT_bf16(W, DD, n0 + 16 * j, k0, lane);
      acc[0][j] = wmma_bf(a0, bj, acc[0][j]);
      acc[1][j] = wmma_bf(a1, bj, acc[1][j]);
    }
  }

  int grp = (lane >> 4) & 1;
  int nlane = lane & 15;
  if (!vt_mode) {
#pragma unroll
    for (int i = 0; i < 2; ++i)
#pragma unroll
      for (int j = 0; j < 4; ++j) {
        int rowb = m0 + 16 * i + grp * 8;
        bf16_t* ob = out + (size_t)rowb * DD + n0 + 16 * j + nlane;
#pragma unroll
        for (int r = 0; r < 8; ++r) ob[(size_t)r * DD] = f2bf(acc[i][j][r]);
      }
  } else {
#pragma unroll
    for (int i = 0; i < 2; ++i)
#pragma unroll
      for (int j = 0; j < 4; ++j) {
        int nc = n0 + 16 * j + nlane;
        int h = nc / DH, dh = nc % DH;
        int rowb = m0 + 16 * i + grp * 8;
        int bb = rowb / SS, s = rowb % SS; // 16-row tile never crosses batch
        bf16_t* vb = out + ((size_t)(bb * HH + h) * DH + dh) * SS + s;
#pragma unroll
        for (int r = 0; r < 8; ++r) vb[r] = f2bf(acc[i][j][r]);
      }
  }
}

// ============================================================
// Kernel 2: softmax row statistics (m, l) per (b,h,row), online, two-level:
// per-lane online update over its column subset, then 16-lane shuffle merge.
// ============================================================
__global__ __launch_bounds__(256) void attn_stats_kernel(
    const bf16_t* __restrict__ Q, const bf16_t* __restrict__ K,
    float* __restrict__ ws_m, float* __restrict__ ws_l) {
  int lane = threadIdx.x & 31;
  int wave = threadIdx.x >> 5;
  int bh = blockIdx.y;
  int bb = bh >> 4, h = bh & 15;
  int qt = blockIdx.x * 8 + wave; // q-tile index, uniform per wave
  int q0 = qt * 16;
  int grp = (lane >> 4) & 1, nlane = lane & 15;

  const bf16_t* pQ = Q + (size_t)bb * SS * DD + h * DH;
  const bf16_t* pK = K + (size_t)bb * SS * DD + h * DH;

  v16bf aq[4];
#pragma unroll
  for (int d = 0; d < 4; ++d) aq[d] = load_a_bf16(pQ, DD, q0, d * 32, lane);

  float mrow[8], lrow[8];
#pragma unroll
  for (int r = 0; r < 8; ++r) {
    mrow[r] = MINIT;
    lrow[r] = 0.0f;
  }

  for (int kt = 0; kt <= qt; ++kt) { // wave-uniform loop (EXEC stays full)
    int k0 = kt * 16;
    v8f sc = (v8f){};
#pragma unroll
    for (int d = 0; d < 4; ++d)
      sc = wmma_bf(aq[d], load_bT_bf16(pK, DD, k0, d * 32, lane), sc);
#pragma unroll
    for (int r = 0; r < 8; ++r) {
      int row = q0 + grp * 8 + r;
      if (k0 + nlane <= row) { // causal
        float s = sc[r];
        if (s > mrow[r]) {
          lrow[r] = lrow[r] * __expf(mrow[r] - s) + 1.0f;
          mrow[r] = s;
        } else {
          lrow[r] += __expf(s - mrow[r]);
        }
      }
    }
  }
  // merge across the 16 lanes holding one row
#pragma unroll
  for (int off = 1; off < 16; off <<= 1) {
#pragma unroll
    for (int r = 0; r < 8; ++r) {
      float mo = __shfl_xor(mrow[r], off, 32);
      float lo = __shfl_xor(lrow[r], off, 32);
      float mn = fmaxf(mrow[r], mo);
      lrow[r] = lrow[r] * __expf(mrow[r] - mn) + lo * __expf(mo - mn);
      mrow[r] = mn;
    }
  }
  if (nlane == 0) {
#pragma unroll
    for (int r = 0; r < 8; ++r) {
      int row = q0 + grp * 8 + r;
      ws_m[(size_t)bh * SS + row] = mrow[r];
      ws_l[(size_t)bh * SS + row] = lrow[r];
    }
  }
}

// ============================================================
// Kernel 3: recompute scores, emit normalized attn_weights (full S x S,
// zeros above the diagonal), and fuse P @ V through an LDS relayout
// (C-fragment f32 -> A-fragment bf16).  Output O (bf16, [B,S,D]).
// ============================================================
__global__ __launch_bounds__(256) void attn_pv_kernel(
    const bf16_t* __restrict__ Q, const bf16_t* __restrict__ K,
    const bf16_t* __restrict__ Vt, const float* __restrict__ ws_m,
    const float* __restrict__ ws_l, float* __restrict__ attnW,
    bf16_t* __restrict__ O) {
  __shared__ bf16_t plds[8][16 * 32]; // one 16x32 P tile per wave (8 KB)
  int lane = threadIdx.x & 31;
  int wave = threadIdx.x >> 5;
  int bh = blockIdx.y;
  int bb = bh >> 4, h = bh & 15;
  int qt = blockIdx.x * 8 + wave;
  int q0 = qt * 16;
  int grp = (lane >> 4) & 1, nlane = lane & 15;

  const bf16_t* pQ = Q + (size_t)bb * SS * DD + h * DH;
  const bf16_t* pK = K + (size_t)bb * SS * DD + h * DH;
  const bf16_t* pV = Vt + (size_t)bh * DH * SS;
  float* pW = attnW + (size_t)bh * SS * SS;
  bf16_t* myLds = &plds[wave][0];

  v16bf aq[4];
#pragma unroll
  for (int d = 0; d < 4; ++d) aq[d] = load_a_bf16(pQ, DD, q0, d * 32, lane);

  float mr[8], ir[8];
#pragma unroll
  for (int r = 0; r < 8; ++r) {
    int row = q0 + grp * 8 + r;
    mr[r] = ws_m[(size_t)bh * SS + row];
    ir[r] = 1.0f / ws_l[(size_t)bh * SS + row];
  }

  v8f o[8];
#pragma unroll
  for (int j = 0; j < 8; ++j) o[j] = (v8f){};

  for (int kk = 0; kk < SS / 32; ++kk) { // pairs of 16-wide key tiles
#pragma unroll
    for (int t = 0; t < 2; ++t) {
      int kt = kk * 2 + t;
      int k0 = kt * 16;
      float pv[8];
      if (kt <= qt) { // uniform branch
        v8f sc = (v8f){};
#pragma unroll
        for (int d = 0; d < 4; ++d)
          sc = wmma_bf(aq[d], load_bT_bf16(pK, DD, k0, d * 32, lane), sc);
#pragma unroll
        for (int r = 0; r < 8; ++r) {
          int row = q0 + grp * 8 + r;
          pv[r] = (k0 + nlane <= row) ? __expf(sc[r] - mr[r]) * ir[r] : 0.0f;
        }
      } else {
#pragma unroll
        for (int r = 0; r < 8; ++r) pv[r] = 0.0f;
      }
      // attn_weights output (fp32), full matrix incl. masked zeros
#pragma unroll
      for (int r = 0; r < 8; ++r)
        pW[(size_t)(q0 + grp * 8 + r) * SS + k0 + nlane] = pv[r];
      // stash bf16 P into this wave's LDS tile (C-layout scatter)
#pragma unroll
      for (int r = 0; r < 8; ++r)
        myLds[(grp * 8 + r) * 32 + t * 16 + nlane] = f2bf(pv[r]);
    }
    if (kk * 2 <= qt) { // any live columns in this 32-wide slab
      // reload P as an A fragment: 8 x ds_load_b32 (DS in-order within wave)
      v16bf pa = load_a_bf16(myLds, 32, 0, 0, lane);
#pragma unroll
      for (int j = 0; j < 8; ++j)
        o[j] = wmma_bf(pa, load_bT_bf16(pV, SS, j * 16, kk * 32, lane), o[j]);
    }
  }

  bf16_t* pO = O + (size_t)bb * SS * DD + h * DH;
#pragma unroll
  for (int j = 0; j < 8; ++j)
#pragma unroll
    for (int r = 0; r < 8; ++r)
      pO[(size_t)(q0 + grp * 8 + r) * DD + j * 16 + nlane] = f2bf(o[j][r]);
}

// ============================================================
// Kernel 4: output projection  out[m,n] = sum_k O[m,k]*Wo[n,k] + bo[n] (fp32)
// ============================================================
__global__ __launch_bounds__(256) void gemm_out_kernel(
    const bf16_t* __restrict__ A, const bf16_t* __restrict__ W,
    const float* __restrict__ bias, float* __restrict__ out) {
  int lane = threadIdx.x & 31;
  int wave = threadIdx.x >> 5;
  int wm = wave >> 2, wn = wave & 3;
  int m0 = blockIdx.y * 64 + wm * 32;
  int n0 = blockIdx.x * 256 + wn * 64;

  v8f acc[2][4];
#pragma unroll
  for (int i = 0; i < 2; ++i)
#pragma unroll
    for (int j = 0; j < 4; ++j) acc[i][j] = (v8f){};

  const bf16_t* aRow0 = A + (size_t)(m0 + (lane & 15)) * DD;
  const bf16_t* bRow0 = W + (size_t)(n0 + (lane & 15)) * DD;

  for (int k0 = 0; k0 < DD; k0 += 32) {
    if (k0 + 32 < DD) {
      __builtin_prefetch(aRow0 + k0 + 32, 0, 3);
      __builtin_prefetch(bRow0 + k0 + 32, 0, 3);
    }
    v16bf a0 = load_a_bf16(A, DD, m0, k0, lane);
    v16bf a1 = load_a_bf16(A, DD, m0 + 16, k0, lane);
#pragma unroll
    for (int j = 0; j < 4; ++j) {
      v16bf bj = load_bT_bf16(W, DD, n0 + 16 * j, k0, lane);
      acc[0][j] = wmma_bf(a0, bj, acc[0][j]);
      acc[1][j] = wmma_bf(a1, bj, acc[1][j]);
    }
  }

  int grp = (lane >> 4) & 1;
  int nlane = lane & 15;
#pragma unroll
  for (int i = 0; i < 2; ++i)
#pragma unroll
    for (int j = 0; j < 4; ++j) {
      int col = n0 + 16 * j + nlane;
      float bv = bias[col];
      int rowb = m0 + 16 * i + grp * 8;
      float* ob = out + (size_t)rowb * DD + col;
#pragma unroll
      for (int r = 0; r < 8; ++r) ob[(size_t)r * DD] = acc[i][j][r] + bv;
    }
}

extern "C" void kernel_launch(void* const* d_in, const int* in_sizes, int n_in,
                              void* d_out, int out_size, void* d_ws,
                              size_t ws_size, hipStream_t stream) {
  const float* X = (const float*)d_in[0];
  const float* wq = (const float*)d_in[1];
  const float* wk = (const float*)d_in[2];
  const float* wv = (const float*)d_in[3];
  const float* wo = (const float*)d_in[4];
  const float* bo = (const float*)d_in[5];

  float* out = (float*)d_out;
  float* attnW = out + (size_t)MM * DD; // attn_weights after attn_output

  // Workspace partition (~193 MB)
  char* w = (char*)d_ws;
  bf16_t* Xb = (bf16_t*)w;  w += (size_t)MM * DD * sizeof(bf16_t);
  bf16_t* Wqb = (bf16_t*)w; w += (size_t)DD * DD * sizeof(bf16_t);
  bf16_t* Wkb = (bf16_t*)w; w += (size_t)DD * DD * sizeof(bf16_t);
  bf16_t* Wvb = (bf16_t*)w; w += (size_t)DD * DD * sizeof(bf16_t);
  bf16_t* Wob = (bf16_t*)w; w += (size_t)DD * DD * sizeof(bf16_t);
  bf16_t* Qb = (bf16_t*)w;  w += (size_t)MM * DD * sizeof(bf16_t);
  bf16_t* Kb = (bf16_t*)w;  w += (size_t)MM * DD * sizeof(bf16_t);
  bf16_t* Vt = (bf16_t*)w;  w += (size_t)BHN * DH * SS * sizeof(bf16_t);
  bf16_t* Ob = (bf16_t*)w;  w += (size_t)MM * DD * sizeof(bf16_t);
  float* ws_m = (float*)w;  w += (size_t)BHN * SS * sizeof(float);
  float* ws_l = (float*)w;

  // One-shot fp32 -> bf16 conversion (amortized across all GEMM re-reads)
  int ngX = (MM * DD) / 8;  // 2,097,152 groups of 8 floats
  int ngW = (DD * DD) / 8;  // 524,288
  cvt_bf16_kernel<<<ngX / 256, 256, 0, stream>>>((const float4*)X, (uint4*)Xb, ngX);
  cvt_bf16_kernel<<<ngW / 256, 256, 0, stream>>>((const float4*)wq, (uint4*)Wqb, ngW);
  cvt_bf16_kernel<<<ngW / 256, 256, 0, stream>>>((const float4*)wk, (uint4*)Wkb, ngW);
  cvt_bf16_kernel<<<ngW / 256, 256, 0, stream>>>((const float4*)wv, (uint4*)Wvb, ngW);
  cvt_bf16_kernel<<<ngW / 256, 256, 0, stream>>>((const float4*)wo, (uint4*)Wob, ngW);

  dim3 gg(DD / 256, MM / 64); // (8, 128)
  gemm_qkv_kernel<<<gg, 256, 0, stream>>>(Xb, Wqb, Qb, 0);
  gemm_qkv_kernel<<<gg, 256, 0, stream>>>(Xb, Wkb, Kb, 0);
  gemm_qkv_kernel<<<gg, 256, 0, stream>>>(Xb, Wvb, Vt, 1);

  dim3 ga(SS / 16 / 8, BHN); // (16, 64)
  attn_stats_kernel<<<ga, 256, 0, stream>>>(Qb, Kb, ws_m, ws_l);
  attn_pv_kernel<<<ga, 256, 0, stream>>>(Qb, Kb, Vt, ws_m, ws_l, attnW, Ob);

  gemm_out_kernel<<<gg, 256, 0, stream>>>(Ob, Wob, bo, out);
}
